// LinearAttention_37288906064643
// MI455X (gfx1250) — matrix operand order repaired and checked by reading, not compiled, not run on hardware
//
#include <hip/hip_runtime.h>
#include <hip/hip_bf16.h>
#include <cstdint>

// ---------- types ----------
typedef unsigned short u16;
typedef __attribute__((ext_vector_type(16))) __bf16 bf16x16;
typedef __attribute__((ext_vector_type(8)))  float  f32x8;
typedef __attribute__((ext_vector_type(4)))  unsigned int u32x4;
typedef __attribute__((ext_vector_type(8)))  int i32x8;
typedef __attribute__((ext_vector_type(4)))  int i32x4;

union Frag16 { bf16x16 v; uint4 q[2]; };

__device__ __forceinline__ u16 f2bf(float f) {
    unsigned int u = __float_as_uint(f);
    unsigned int r = (u + 0x7FFFu + ((u >> 16) & 1u)) >> 16;
    return (u16)r;
}

// A fragment: rows m0..m0+15, cols k0..k0+31 of row-major A (ld elements per row).
__device__ __forceinline__ bf16x16 load_frag_a(const u16* base, int ld, int m0, int k0, int lane) {
    int r = lane & 15, h = lane >> 4;
    const u16* p = base + (size_t)(m0 + r) * ld + k0 + 8 * h;
    Frag16 f;
    f.q[0] = *(const uint4*)(p);
    f.q[1] = *(const uint4*)(p + 16);
    return f.v;
}

// B fragment from B^T storage: BT[n][k] row-major (ld elems). Cols n0..n0+15, K k0..k0+31.
__device__ __forceinline__ bf16x16 load_frag_b(const u16* baseT, int ld, int n0, int k0, int lane) {
    int r = lane & 15, h = lane >> 4;
    const u16* p = baseT + (size_t)(n0 + r) * ld + k0 + 8 * h;
    Frag16 f;
    f.q[0] = *(const uint4*)(p);
    f.q[1] = *(const uint4*)(p + 16);
    return f.v;
}

__device__ __forceinline__ f32x8 wmma_bf16(bf16x16 a, bf16x16 b, f32x8 c) {
    return __builtin_amdgcn_wmma_f32_16x16x32_bf16(false, a, false, b, (short)0, c, false, false);
}

// ---------- Tensor Data Mover: 2D tile Global -> LDS (bf16 elements) ----------
// D# packing per CDNA5 ISA ch.8 (group0 128b, group1 256b, groups2/3 zero for 2D).
__device__ __forceinline__ void tdm_load_2d(unsigned lds_off, const void* gaddr,
                                            unsigned tile_d0, unsigned tile_d1,
                                            unsigned tensor_d0, unsigned tensor_d1,
                                            unsigned long long stride0) {
    unsigned long long ga = (unsigned long long)(uintptr_t)gaddr;
    u32x4 g0;
    g0[0] = 1u;                                   // count=1 (valid user descriptor)
    g0[1] = lds_off;                              // LDS byte address
    g0[2] = (unsigned)(ga & 0xFFFFFFFFu);         // global_addr[31:0]
    g0[3] = (unsigned)((ga >> 32) & 0x01FFFFFFu)  // global_addr[56:32]
          | (2u << 30);                           // type = 2 ("image")
    i32x8 g1;
    g1[0] = (int)(1u << 16);                      // wg_mask=0, data_size=1 (2 bytes)
    g1[1] = (int)((tensor_d0 & 0xFFFFu) << 16);   // tensor_dim0[15:0]
    g1[2] = (int)(((tensor_d0 >> 16) & 0xFFFFu) | ((tensor_d1 & 0xFFFFu) << 16));
    g1[3] = (int)(((tensor_d1 >> 16) & 0xFFFFu) | ((tile_d0 & 0xFFFFu) << 16));
    g1[4] = (int)(tile_d1 & 0xFFFFu);             // tile_dim1; tile_dim2=0
    g1[5] = (int)(unsigned)(stride0 & 0xFFFFFFFFull);       // tensor_dim0_stride[31:0]
    g1[6] = (int)(unsigned)((stride0 >> 32) & 0xFFFFull);   // stride[47:32]; dim1_stride=0
    g1[7] = 0;
    i32x4 g2 = {0, 0, 0, 0};
    i32x4 g3 = {0, 0, 0, 0};
#if __clang_major__ >= 23
    i32x8 g4 = {0, 0, 0, 0, 0, 0, 0, 0};
    __builtin_amdgcn_tensor_load_to_lds(g0, g1, g2, g3, g4, 0);
#else
    __builtin_amdgcn_tensor_load_to_lds(g0, g1, g2, g3, 0);
#endif
}

__device__ __forceinline__ unsigned lds_offset(const void* p) {
    return (unsigned)(uintptr_t)p;   // flat LDS address low 32 bits = LDS byte offset
}

// ---------- prep kernels ----------
__global__ void k_convert_bf16(const float* __restrict__ src, u16* __restrict__ dst, int n) {
    int i = blockIdx.x * blockDim.x + threadIdx.x;
    if (i < n) dst[i] = f2bf(src[i]);
}

__global__ void k_transpose_bf16(const float* __restrict__ src, u16* __restrict__ dst, int rows, int cols) {
    int i = blockIdx.x * blockDim.x + threadIdx.x;
    if (i < rows * cols) {
        int r = i / cols, c = i % cols;
        dst[(size_t)c * rows + r] = f2bf(src[i]);
    }
}

// dstT[n*512 + d] = bf16( sum_t W[d*512 + t] * A[t*512 + n] )  -> (W@A)^T in bf16
__global__ void k_fuse_weight(const float* __restrict__ W, const float* __restrict__ A, u16* __restrict__ dstT) {
    int i = blockIdx.x * blockDim.x + threadIdx.x;   // over 512*512
    int n = i >> 9, d = i & 511;
    float s = 0.f;
    for (int t = 0; t < 512; ++t) s += W[d * 512 + t] * A[t * 512 + n];
    dstT[(size_t)n * 512 + d] = f2bf(s);
}

// ---------- generic WMMA GEMM ----------
// C[M,N] = A[M,K] bf16 row-major  x  BT[N,K] bf16. Block tile 128x128, wave tile 32x64.
// Output modes: Cbf (bf16, scaled) | CbfT (bf16 transposed [N][ldt], scaled) | Cf (f32 + bias).
__global__ void __launch_bounds__(256)
k_gemm(const u16* __restrict__ Aw, const u16* __restrict__ BT,
       u16* __restrict__ Cbf, u16* __restrict__ CbfT, float* __restrict__ Cf,
       const float* __restrict__ bias, float scale, int N, int K, int ldt) {
    const int lane = threadIdx.x & 31;
    const int wid  = threadIdx.x >> 5;
    const int m0 = blockIdx.x * 128 + (wid >> 1) * 32;
    const int n0 = blockIdx.y * 128 + (wid & 1) * 64;

    f32x8 acc[2][4];
    f32x8 z = {};
#pragma unroll
    for (int mt = 0; mt < 2; ++mt)
#pragma unroll
        for (int nt = 0; nt < 4; ++nt) acc[mt][nt] = z;

    for (int k = 0; k < K; k += 32) {
        bf16x16 a0 = load_frag_a(Aw, K, m0,      k, lane);
        bf16x16 a1 = load_frag_a(Aw, K, m0 + 16, k, lane);
#pragma unroll
        for (int nt = 0; nt < 4; ++nt) {
            bf16x16 b = load_frag_b(BT, K, n0 + 16 * nt, k, lane);
            acc[0][nt] = wmma_bf16(a0, b, acc[0][nt]);
            acc[1][nt] = wmma_bf16(a1, b, acc[1][nt]);
        }
    }

    const int cr = lane & 15, h = lane >> 4;
    if (CbfT) {
        // transposed bf16 store: per lane, 8 consecutive rows of one column -> one 16B store
#pragma unroll
        for (int mt = 0; mt < 2; ++mt)
#pragma unroll
            for (int nt = 0; nt < 4; ++nt) {
                int col = n0 + 16 * nt + cr;
                int rowbase = m0 + 16 * mt + 8 * h;
                u16 tmp[8];
#pragma unroll
                for (int j = 0; j < 8; ++j) tmp[j] = f2bf(acc[mt][nt][j] * scale);
                *(uint4*)(CbfT + (size_t)col * ldt + rowbase) = *(const uint4*)tmp;
            }
    } else {
#pragma unroll
        for (int mt = 0; mt < 2; ++mt)
#pragma unroll
            for (int nt = 0; nt < 4; ++nt)
#pragma unroll
                for (int j = 0; j < 8; ++j) {
                    int row = m0 + 16 * mt + j + 8 * h;
                    int col = n0 + 16 * nt + cr;
                    float v = acc[mt][nt][j] * scale;
                    if (Cbf) Cbf[(size_t)row * N + col] = f2bf(v);
                    else     Cf [(size_t)row * N + col] = v + bias[col];
                }
    }
}

// ---------- flash attention with TDM double-buffered K/V tiles ----------
// Grid: (S/64, B). WG = 256 threads (8 waves) owns 64 query rows, streams 32-key chunks.
// P phase: wave w computes P[16*(w&3)..+16, 16*(w>>2)..+16]. O phase: wave w owns cols w*64..+64.
// Wave 0 DMAs next K tile, wave 1 next V tile (double buffer), wave 2 DMAs Q once.
__global__ void __launch_bounds__(256)
k_flash(const u16* __restrict__ Qb, const u16* __restrict__ Kb,
        const u16* __restrict__ VbT, u16* __restrict__ Ob) {
    extern __shared__ char smem_raw[];
    u16*   Qs    = (u16*)smem_raw;              // [64][512]
    u16*   Ks0   = Qs  + 64 * 512;              // [32][512] keys row-major, buffer 0
    u16*   Ks1   = Ks0 + 32 * 512;              // buffer 1
    u16*   Vt0   = Ks1 + 32 * 512;              // [512][32] V^T tile, buffer 0
    u16*   Vt1   = Vt0 + 32 * 512;              // buffer 1
    float* Pf    = (float*)(Vt1 + 32 * 512);    // [64][32] logits
    u16*   Pb    = (u16*)(Pf + 64 * 32);        // [64][32] exp() bf16
    float* rowM  = (float*)(Pb + 64 * 32);      // [64]
    float* rowL  = rowM + 64;
    float* rowC  = rowL + 64;
    u16* const KsB[2] = {Ks0, Ks1};
    u16* const VtB[2] = {Vt0, Vt1};

    const int tid  = threadIdx.x;
    const int lane = tid & 31;
    const int wid  = tid >> 5;
    const int b    = blockIdx.y;
    const int q0   = blockIdx.x * 64;

    const u16* Kbase = Kb  + (size_t)b * 4096 * 512;   // [4096][512]
    const u16* Vbase = VbT + (size_t)b * 4096;         // [512][16384], batch col offset

    // prologue: DMA Q tile + chunk-0 K/V tiles
    if (wid == 0)      tdm_load_2d(lds_offset(Ks0), Kbase, 512, 32, 512, 4096, 512);
    else if (wid == 1) tdm_load_2d(lds_offset(Vt0), Vbase, 32, 512, 16384, 512, 16384);
    else if (wid == 2) tdm_load_2d(lds_offset(Qs), Qb + (size_t)(b * 4096 + q0) * 512,
                                   512, 64, 512, 4096, 512);
    if (tid < 64) { rowM[tid] = -1e30f; rowL[tid] = 0.f; }
    if (wid < 3) __builtin_amdgcn_s_wait_tensorcnt(0);
    __syncthreads();

    f32x8 acc[4][4];
    f32x8 z = {};
#pragma unroll
    for (int mt = 0; mt < 4; ++mt)
#pragma unroll
        for (int nt = 0; nt < 4; ++nt) acc[mt][nt] = z;

    const int pr0 = 16 * (wid & 3);
    const int pj0 = 16 * (wid >> 2);

    for (int it = 0; it < 128; ++it) {
        const int cur = it & 1;
        // kick off DMA for the next chunk into the other buffer (overlaps with compute)
        if (it + 1 < 128) {
            const int cn = (it + 1) * 32;
            if (wid == 0)
                tdm_load_2d(lds_offset(KsB[cur ^ 1]), Kbase + (size_t)cn * 512,
                            512, 32, 512, 4096, 512);
            else if (wid == 1)
                tdm_load_2d(lds_offset(VtB[cur ^ 1]), Vbase + cn,
                            32, 512, 16384, 512, 16384);
        }

        // P = Q @ K^T sub-tile (16x16 per wave), K-dim = 512
        const u16* Kc = KsB[cur];
        f32x8 pacc = z;
        for (int k = 0; k < 512; k += 32) {
            bf16x16 a  = load_frag_a(Qs, 512, pr0, k, lane);
            bf16x16 bb = load_frag_b(Kc, 512, pj0, k, lane);
            pacc = wmma_bf16(a, bb, pacc);
        }
        {
            int h = lane >> 4, cr = lane & 15;
#pragma unroll
            for (int j = 0; j < 8; ++j)
                Pf[(pr0 + j + 8 * h) * 32 + pj0 + cr] = pacc[j];
        }
        __syncthreads();

        // online softmax: one thread per query row
        if (tid < 64) {
            int r = tid;
            float mo = rowM[r], mx = mo;
            for (int j2 = 0; j2 < 32; ++j2) mx = fmaxf(mx, Pf[r * 32 + j2]);
            float corr = __expf(mo - mx);
            float s = 0.f;
            for (int j2 = 0; j2 < 32; ++j2) {
                float e = __expf(Pf[r * 32 + j2] - mx);
                Pb[r * 32 + j2] = f2bf(e);
                s += e;
            }
            rowL[r] = rowL[r] * corr + s;
            rowM[r] = mx;
            rowC[r] = corr;
        }
        __syncthreads();

        // rescale accumulators, then O += exp(P) @ V
        {
            int h = lane >> 4;
#pragma unroll
            for (int mt = 0; mt < 4; ++mt) {
                float cs[8];
#pragma unroll
                for (int j = 0; j < 8; ++j) cs[j] = rowC[16 * mt + j + 8 * h];
#pragma unroll
                for (int nt = 0; nt < 4; ++nt)
#pragma unroll
                    for (int j = 0; j < 8; ++j) acc[mt][nt][j] *= cs[j];
            }
        }
        {
            const u16* Vc = VtB[cur];
            bf16x16 af[4];
#pragma unroll
            for (int mt = 0; mt < 4; ++mt) af[mt] = load_frag_a(Pb, 32, 16 * mt, 0, lane);
#pragma unroll
            for (int nt = 0; nt < 4; ++nt) {
                bf16x16 bb = load_frag_b(Vc, 32, wid * 64 + 16 * nt, 0, lane);
#pragma unroll
                for (int mt = 0; mt < 4; ++mt)
                    acc[mt][nt] = wmma_bf16(af[mt], bb, acc[mt][nt]);
            }
        }

        // drain next chunk's DMA before anyone touches the other buffer
        if (wid < 2) __builtin_amdgcn_s_wait_tensorcnt(0);
        __syncthreads();
    }

    // normalize by rowL and write bf16 output
    {
        int h = lane >> 4, cr = lane & 15;
#pragma unroll
        for (int mt = 0; mt < 4; ++mt) {
            float li[8];
#pragma unroll
            for (int j = 0; j < 8; ++j) li[j] = 1.f / rowL[16 * mt + j + 8 * h];
#pragma unroll
            for (int nt = 0; nt < 4; ++nt)
#pragma unroll
                for (int j = 0; j < 8; ++j) {
                    int row = 16 * mt + j + 8 * h;
                    int d   = wid * 64 + 16 * nt + cr;
                    Ob[(size_t)(b * 4096 + q0 + row) * 512 + d] = f2bf(acc[mt][nt][j] * li[j]);
                }
        }
    }
}

// ---------- launch ----------
extern "C" void kernel_launch(void* const* d_in, const int* in_sizes, int n_in,
                              void* d_out, int out_size, void* d_ws, size_t ws_size,
                              hipStream_t stream) {
    const float* x  = (const float*)d_in[0];
    const float* Wq = (const float*)d_in[1];
    const float* Wk = (const float*)d_in[2];
    const float* Wv = (const float*)d_in[3];
    const float* Am = (const float*)d_in[4];
    const float* Wo = (const float*)d_in[5];
    const float* bo = (const float*)d_in[6];
    float* out = (float*)d_out;

    const size_t BS  = (size_t)4 * 4096;      // 16384 rows
    const size_t BIG = BS * 512;              // elements per [B*S,512] buffer
    u16* xb   = (u16*)d_ws;
    u16* Qb   = xb  + BIG;
    u16* Khb  = Qb  + BIG;
    u16* VhbT = Khb + BIG;                    // V_hat^T: [512][16384]
    u16* Ob   = VhbT + BIG;
    u16* WqT  = Ob  + BIG;
    u16* WkaT = WqT  + 512 * 512;
    u16* WvaT = WkaT + 512 * 512;
    u16* WoT  = WvaT + 512 * 512;

    // prep
    k_convert_bf16<<<(int)(BIG / 256), 256, 0, stream>>>(x, xb, (int)BIG);
    k_transpose_bf16<<<1024, 256, 0, stream>>>(Wq, WqT, 512, 512);
    k_transpose_bf16<<<1024, 256, 0, stream>>>(Wo, WoT, 512, 512);
    k_fuse_weight<<<1024, 256, 0, stream>>>(Wk, Am, WkaT);
    k_fuse_weight<<<1024, 256, 0, stream>>>(Wv, Am, WvaT);

    // projections: Q (pre-scaled by 1/sqrt(d)), K_hat, V_hat^T
    dim3 gg(16384 / 128, 512 / 128);
    const float qscale = 0.044194173824159216f;  // 1/sqrt(512)
    k_gemm<<<gg, 256, 0, stream>>>(xb, WqT,  Qb,  nullptr, nullptr, nullptr, qscale, 512, 512, 0);
    k_gemm<<<gg, 256, 0, stream>>>(xb, WkaT, Khb, nullptr, nullptr, nullptr, 1.0f,   512, 512, 0);
    k_gemm<<<gg, 256, 0, stream>>>(xb, WvaT, nullptr, VhbT, nullptr, nullptr, 1.0f,  512, 512, 16384);

    // flash attention (dynamic LDS: Q 64K + 2xK 32K + 2xV 32K + P bufs + row state)
    size_t shmem = (size_t)64 * 512 * 2        // Qs
                 + (size_t)2 * 32 * 512 * 2    // Ks0/Ks1
                 + (size_t)2 * 32 * 512 * 2    // Vt0/Vt1
                 + (size_t)64 * 32 * 4         // Pf
                 + (size_t)64 * 32 * 2         // Pb
                 + (size_t)3 * 64 * 4;         // rowM/L/C
    k_flash<<<dim3(64, 4), 256, shmem, stream>>>(Qb, Khb, VhbT, Ob);

    // output projection (f32 + bias)
    k_gemm<<<gg, 256, 0, stream>>>(Ob, WoT, nullptr, nullptr, out, bo, 1.0f, 512, 512, 0);
}
